// GaussMemoryStep_71347996721428
// MI455X (gfx1250) — compile-verified, matrix operand order checked by reading, not run on hardware
//
#include <hip/hip_runtime.h>

// ---------------------------------------------------------------------------
// GaussMemoryStep for gfx1250 (MI455X): the band-limited FFT projections are
// folded into precomputed dense bases, so the whole pipeline is bf16 WMMA
// GEMMs (v_wmma_f32_16x16x32_bf16) with f32 accumulation.
//   WallT [768,8192]  : fused q/k/v projection basis (N-major, K-contig)
//   WoT   [8192,256]  : synthesis basis (N-major, K-contig)
//   q,k   [B*T,256]   bf16 ;  vT [B,256,T] bf16 (transposed at store time)
//   scores[B,T,T]     bf16 (decay mask fused) ;  R [B*T,256] bf16
// Large GEMMs use 32x32 register tiles (2x2 WMMA) per wave; grids are ordered
// so the streamed operand (x / out) passes through L2 exactly once.
// ---------------------------------------------------------------------------

typedef __attribute__((ext_vector_type(16))) __bf16 v16bf;
typedef __attribute__((ext_vector_type(8)))  float  v8f;

#define NB  4
#define NT  2048
#define NC  256
#define NV  8192
#define NF  256
#define NBT (NB * NT)

#define TWOPI_OVER_V 7.66990393942820598e-4f   // 2*pi/8192
#define IRFFT_SCALE  2.44140625e-4f            // 2/8192

__device__ __forceinline__ unsigned short f2bf(float f) {
  union { float f; unsigned int u; } c; c.f = f;
  unsigned int u = c.u;
  u += 0x7fffu + ((u >> 16) & 1u);             // round-to-nearest-even
  return (unsigned short)(u >> 16);
}

union FragU {
  v16bf v;
  unsigned short u[16];
  uint4 q[2];
};

// bf16 fragment, source row-major [rows, ld], K contiguous.
// Works for A (row = M) and for B when the source is stored N-major.
// ISA 16-bit 16x32 layout: lanes 0-15 K{0..7,16..23}, lanes 16-31 K{8..15,24..31}.
__device__ __forceinline__ v16bf load_frag_bf16(const unsigned short* __restrict__ base,
                                                int ld, int row0, int k0, int lane) {
  const int r  = lane & 15;
  const int ko = (lane >> 4) << 3;
  const unsigned short* p = base + (size_t)(row0 + r) * ld + (size_t)(k0 + ko);
  FragU f;
  f.q[0] = *(const uint4*)(p);
  f.q[1] = *(const uint4*)(p + 16);
  return f.v;
}

// A-fragment from an f32 source (x), converted to bf16 on the fly.
__device__ __forceinline__ v16bf load_frag_f32(const float* __restrict__ base,
                                               int ld, int row0, int k0, int lane) {
  const int r  = lane & 15;
  const int ko = (lane >> 4) << 3;
  const float* p = base + (size_t)(row0 + r) * ld + (size_t)(k0 + ko);
  float4 a0 = *(const float4*)(p);
  float4 a1 = *(const float4*)(p + 4);
  float4 b0 = *(const float4*)(p + 16);
  float4 b1 = *(const float4*)(p + 20);
  FragU f;
  f.u[0]  = f2bf(a0.x); f.u[1]  = f2bf(a0.y); f.u[2]  = f2bf(a0.z); f.u[3]  = f2bf(a0.w);
  f.u[4]  = f2bf(a1.x); f.u[5]  = f2bf(a1.y); f.u[6]  = f2bf(a1.z); f.u[7]  = f2bf(a1.w);
  f.u[8]  = f2bf(b0.x); f.u[9]  = f2bf(b0.y); f.u[10] = f2bf(b0.z); f.u[11] = f2bf(b0.w);
  f.u[12] = f2bf(b1.x); f.u[13] = f2bf(b1.y); f.u[14] = f2bf(b1.z); f.u[15] = f2bf(b1.w);
  return f.v;
}

__device__ __forceinline__ v8f wmma_bf16(v16bf a, v16bf b, v8f c) {
  return __builtin_amdgcn_wmma_f32_16x16x32_bf16(false, a, false, b, (short)0, c,
                                                 false, false);
}

// --------------------------- basis precompute ------------------------------

// WallT[n, v] = sum_f w[c,f]*cos(2pi f v/V) - w[c,256+f]*sin(...), f = 1..256
// n: 0-255 -> q_w, 256-511 -> k_w, 512-767 -> v_w
__global__ void gm_build_wall(const float* __restrict__ qw,
                              const float* __restrict__ kw,
                              const float* __restrict__ vw,
                              unsigned short* __restrict__ wallT) {
  const int idx = blockIdx.x * 256 + threadIdx.x;        // n*8192 + v
  const int n = idx >> 13;
  const int v = idx & (NV - 1);
  const float* w = (n < 256) ? qw : ((n < 512) ? kw : vw);
  const int c = n & 255;
  float acc = 0.0f;
  for (int f = 1; f <= NF; ++f) {
    const int ph = (v * f) & (NV - 1);                   // exact range reduction
    float sn, cs;
    __sincosf(TWOPI_OVER_V * (float)ph, &sn, &cs);
    acc += w[c * 512 + (f - 1)] * cs - w[c * 512 + 256 + (f - 1)] * sn;
  }
  wallT[idx] = f2bf(acc);
}

// WoT[v, c] = (2/V) * sum_f ow[c,f]*cos - ow[c,256+f]*sin
__global__ void gm_build_wo(const float* __restrict__ ow,
                            unsigned short* __restrict__ woT) {
  const int idx = blockIdx.x * 256 + threadIdx.x;        // v*256 + c
  const int v = idx >> 8;
  const int c = idx & 255;
  float acc = 0.0f;
  for (int f = 1; f <= NF; ++f) {
    const int ph = (v * f) & (NV - 1);
    float sn, cs;
    __sincosf(TWOPI_OVER_V * (float)ph, &sn, &cs);
    acc += ow[c * 512 + (f - 1)] * cs - ow[c * 512 + 256 + (f - 1)] * sn;
  }
  woT[idx] = f2bf(acc * IRFFT_SCALE);
}

// ------------------------------- GEMMs -------------------------------------

// Store one 16x16 f32 D-tile into the q / k / vT regions (bf16).
__device__ __forceinline__ void gm_store_proj(v8f acc, int mt, int nt, int lane,
                                              unsigned short* __restrict__ qo,
                                              unsigned short* __restrict__ ko,
                                              unsigned short* __restrict__ vT) {
  const int n = nt + (lane & 15);
  const int mbase = mt + ((lane >> 4) << 3);
  if (nt < 256) {
    for (int i = 0; i < 8; ++i) qo[(size_t)(mbase + i) * NC + n] = f2bf(acc[i]);
  } else if (nt < 512) {
    for (int i = 0; i < 8; ++i) ko[(size_t)(mbase + i) * NC + (n - 256)] = f2bf(acc[i]);
  } else {
    const int c = n - 512;
    for (int i = 0; i < 8; ++i) {
      const int m = mbase + i;
      const int bb = m >> 11;
      const int t  = m & (NT - 1);
      vT[((size_t)bb * NC + c) * NT + t] = f2bf(acc[i]);   // transposed store
    }
  }
}

// qkv = x @ WallT^T : M=8192 (b*t), N=768 (q|k|v channels), K=8192 (vocab)
// 32x32 tile per wave; grid.x = N tiles so consecutive blocks reuse x rows
// (x streams through L2 once; 12MB WallT stays L2-resident).
__global__ void gm_proj_gemm(const float* __restrict__ x,
                             const unsigned short* __restrict__ wallT,
                             unsigned short* __restrict__ qo,
                             unsigned short* __restrict__ ko,
                             unsigned short* __restrict__ vT) {
  const int lane = threadIdx.x;
  const int n0 = blockIdx.x * 32;
  const int m0 = blockIdx.y * 32;
  v8f a00 = {}, a01 = {}, a10 = {}, a11 = {};
  for (int kb = 0; kb < NV; kb += 32) {
    v16bf A0 = load_frag_f32(x, NV, m0,      kb, lane);
    v16bf A1 = load_frag_f32(x, NV, m0 + 16, kb, lane);
    v16bf B0 = load_frag_bf16(wallT, NV, n0,      kb, lane);
    v16bf B1 = load_frag_bf16(wallT, NV, n0 + 16, kb, lane);
    __builtin_prefetch(x + (size_t)(m0 + (lane & 15)) * NV + (kb + 32), 0, 1);
    a00 = wmma_bf16(A0, B0, a00);
    a01 = wmma_bf16(A0, B1, a01);
    a10 = wmma_bf16(A1, B0, a10);
    a11 = wmma_bf16(A1, B1, a11);
  }
  gm_store_proj(a00, m0,      n0,      lane, qo, ko, vT);
  gm_store_proj(a01, m0,      n0 + 16, lane, qo, ko, vT);
  gm_store_proj(a10, m0 + 16, n0,      lane, qo, ko, vT);
  gm_store_proj(a11, m0 + 16, n0 + 16, lane, qo, ko, vT);
}

// scores[b,t,s] = decay_weight(t,s) * q.k : per-batch M=N=2048, K=256
__global__ void gm_scores_gemm(const unsigned short* __restrict__ q,
                               const unsigned short* __restrict__ k,
                               unsigned short* __restrict__ s,
                               const float* __restrict__ decay_logit) {
  const int lane = threadIdx.x;
  const int t0 = blockIdx.x * 16;
  const int s0 = blockIdx.y * 16;
  const int b  = blockIdx.z;
  unsigned short* sb = s + (size_t)b * NT * NT;
  const int scol  = s0 + (lane & 15);
  const int tbase = t0 + ((lane >> 4) << 3);
  if (s0 + 15 <= t0) {                                    // strictly lower/diag tile: all zero
    for (int i = 0; i < 8; ++i) sb[(size_t)(tbase + i) * NT + scol] = 0;
    return;
  }
  const unsigned short* qb = q + (size_t)b * NT * NC;
  const unsigned short* kb = k + (size_t)b * NT * NC;
  v8f acc = {};
  for (int c = 0; c < NC; c += 32) {
    v16bf a  = load_frag_bf16(qb, NC, t0, c, lane);
    v16bf bf = load_frag_bf16(kb, NC, s0, c, lane);       // k row-major == B^T, K-contig
    acc = wmma_bf16(a, bf, acc);
  }
  const float dl = decay_logit[0];
  const float decay = 1.0f / (1.0f + __expf(-dl));
  const float l2d = __log2f(decay);
  for (int i = 0; i < 8; ++i) {
    const int t = tbase + i;
    const int d = scol - t;                               // diff = s - t
    const float w = (d > 0) ? exp2f(l2d * (float)(d - 1)) : 0.0f;
    sb[(size_t)t * NT + scol] = f2bf(acc[i] * w);
  }
}

// R = scores @ v : per-batch M=2048, N=256, K=2048. 32x32 tile per wave,
// skipping K tiles that lie entirely in the zero (s <= t) triangle.
__global__ void gm_retr_gemm(const unsigned short* __restrict__ s,
                             const unsigned short* __restrict__ vT,
                             unsigned short* __restrict__ r) {
  const int lane = threadIdx.x;
  const int c0 = blockIdx.x * 32;
  const int t0 = blockIdx.y * 32;
  const int b  = blockIdx.z;
  const unsigned short* sb = s  + (size_t)b * NT * NT;
  const unsigned short* vb = vT + (size_t)b * NC * NT;
  const int kb0 = (t0 >= 31) ? (((t0 - 31) >> 5) << 5) : 0;  // safe: skips only zero tiles
  v8f a00 = {}, a01 = {}, a10 = {}, a11 = {};
  for (int kb = kb0; kb < NT; kb += 32) {
    v16bf A0 = load_frag_bf16(sb, NT, t0,      kb, lane);
    v16bf A1 = load_frag_bf16(sb, NT, t0 + 16, kb, lane);
    v16bf B0 = load_frag_bf16(vb, NT, c0,      kb, lane);  // vT is N-major, K-contig
    v16bf B1 = load_frag_bf16(vb, NT, c0 + 16, kb, lane);
    a00 = wmma_bf16(A0, B0, a00);
    a01 = wmma_bf16(A0, B1, a01);
    a10 = wmma_bf16(A1, B0, a10);
    a11 = wmma_bf16(A1, B1, a11);
  }
  const int rb = (int)b * NT;
  for (int mi = 0; mi < 2; ++mi) {
    const int tbase = t0 + mi * 16 + ((lane >> 4) << 3);
    v8f accL = mi ? a10 : a00;
    v8f accH = mi ? a11 : a01;
    const int cL = c0 + (lane & 15);
    const int cH = cL + 16;
    for (int i = 0; i < 8; ++i) {
      r[((size_t)(rb + tbase + i)) * NC + cL] = f2bf(accL[i]);
      r[((size_t)(rb + tbase + i)) * NC + cH] = f2bf(accH[i]);
    }
  }
}

// out = (R @ WoT^T) * out_scale : M=8192, N=8192, K=256, f32 output.
// 32x32 tile per wave; both operands (4MB each) are L2-resident, the 256MB
// output write is the (unavoidable) HBM stream.
__global__ void gm_synth_gemm(const unsigned short* __restrict__ r,
                              const unsigned short* __restrict__ woT,
                              float* __restrict__ out,
                              const float* __restrict__ out_scale) {
  const int lane = threadIdx.x;
  const int n0 = blockIdx.x * 32;
  const int m0 = blockIdx.y * 32;
  v8f a00 = {}, a01 = {}, a10 = {}, a11 = {};
  for (int kb = 0; kb < NC; kb += 32) {
    v16bf A0 = load_frag_bf16(r,   NC, m0,      kb, lane);
    v16bf A1 = load_frag_bf16(r,   NC, m0 + 16, kb, lane);
    v16bf B0 = load_frag_bf16(woT, NC, n0,      kb, lane);
    v16bf B1 = load_frag_bf16(woT, NC, n0 + 16, kb, lane);
    __builtin_prefetch(woT + (size_t)(n0 + 32 + (lane & 15)) * NC + kb, 0, 1);
    a00 = wmma_bf16(A0, B0, a00);
    a01 = wmma_bf16(A0, B1, a01);
    a10 = wmma_bf16(A1, B0, a10);
    a11 = wmma_bf16(A1, B1, a11);
  }
  const float sc = out_scale[0];
  for (int mi = 0; mi < 2; ++mi) {
    const int mbase = m0 + mi * 16 + ((lane >> 4) << 3);
    v8f accL = mi ? a10 : a00;
    v8f accH = mi ? a11 : a01;
    const int nL = n0 + (lane & 15);
    const int nH = nL + 16;
    for (int i = 0; i < 8; ++i) {
      out[(size_t)(mbase + i) * NV + nL] = accL[i] * sc;
      out[(size_t)(mbase + i) * NV + nH] = accH[i] * sc;
    }
  }
}

// ------------------------------- launch ------------------------------------

extern "C" void kernel_launch(void* const* d_in, const int* in_sizes, int n_in,
                              void* d_out, int out_size, void* d_ws, size_t ws_size,
                              hipStream_t stream) {
  (void)in_sizes; (void)n_in; (void)out_size; (void)ws_size;
  const float* x  = (const float*)d_in[0];
  const float* qw = (const float*)d_in[1];
  const float* kw = (const float*)d_in[2];
  const float* vw = (const float*)d_in[3];
  const float* ow = (const float*)d_in[4];
  const float* dl = (const float*)d_in[5];
  const float* os = (const float*)d_in[6];
  float* out = (float*)d_out;

  char* ws = (char*)d_ws;                         // 64 MB total, 256B-aligned regions
  unsigned short* wall = (unsigned short*)(ws + 0);          // 768*8192*2  = 12 MB
  unsigned short* wo   = (unsigned short*)(ws + 12582912);   // 8192*256*2  =  4 MB
  unsigned short* q    = (unsigned short*)(ws + 16777216);   //               4 MB
  unsigned short* k    = (unsigned short*)(ws + 20971520);   //               4 MB
  unsigned short* vt   = (unsigned short*)(ws + 25165824);   //               4 MB
  unsigned short* sc   = (unsigned short*)(ws + 29360128);   // 4*2048^2*2  = 32 MB
  unsigned short* r    = (unsigned short*)(ws + 62914560);   //               4 MB

  gm_build_wall<<<dim3((768 * NV) / 256), dim3(256), 0, stream>>>(qw, kw, vw, wall);
  gm_build_wo  <<<dim3((NV * NC) / 256),  dim3(256), 0, stream>>>(ow, wo);
  gm_proj_gemm <<<dim3(768 / 32, NBT / 32),      dim3(32), 0, stream>>>(x, wall, q, k, vt);
  gm_scores_gemm<<<dim3(NT / 16, NT / 16, NB),   dim3(32), 0, stream>>>(q, k, sc, dl);
  gm_retr_gemm <<<dim3(NC / 32, NT / 32, NB),    dim3(32), 0, stream>>>(sc, vt, r);
  gm_synth_gemm<<<dim3(NV / 32, NBT / 32),       dim3(32), 0, stream>>>(r, wo, out, os);
}
